// SimpleBEVNetworkWithResNet_33474975105051
// MI455X (gfx1250) — matrix operand order repaired and checked by reading, not compiled, not run on hardware
//
#include <hip/hip_runtime.h>
#include <hip/hip_bf16.h>

typedef __attribute__((ext_vector_type(16))) _Float16 v16h;
typedef __attribute__((ext_vector_type(8)))  float    v8f;
typedef __attribute__((ext_vector_type(4)))  float    f32x4;

#define GH   200
#define GW   200
#define NZV  6
#define CIMG 3
#define IH   1080
#define IW   1920
#define CMID 128
#define KTOT 1152   /* 128 * 9 taps */
#define NB   2

union V16U { v16h h; f32x4 f[2]; };
union H8U  { f32x4 f; _Float16 h[8]; };

// ---------------- Stage A: projection + distortion + bilinear sampling ----
__global__ void k_sample(const float* __restrict__ img,
                         const float* __restrict__ poly,
                         const float* __restrict__ cam,
                         float* __restrict__ xbev) {
  int t = blockIdx.x * blockDim.x + threadIdx.x;
  if (t >= NB * NZV * GH * GW) return;
  int gw = t % GW; int r = t / GW;
  int gh = r % GH; r /= GH;
  int z  = r % NZV; int b = r / NZV;
  // bev = flip(swapaxes(samp,-1,-2)) => sample grid index (GW-1-gw, GH-1-gh)
  float wx = (float)(GW - 1 - gw) * 0.5f - 50.0f;
  float wy = (float)(GH - 1 - gh) * 0.5f - 50.0f;
  float wz = 0.6f * (float)z;                // linspace(0,3,6)
  const float* c = cam + b * 12;
  float p0 = c[0]*wx + c[1]*wy + c[2]*wz  + c[3];
  float p1 = c[4]*wx + c[5]*wy + c[6]*wz  + c[7];
  float p2 = c[8]*wx + c[9]*wy + c[10]*wz + c[11];
  float zc = (fabsf(p2) < 1e-6f) ? 1e-6f : p2;
  float gx = p0 / zc, gy = p1 / zc;
  float rr = sqrtf(gx*gx + gy*gy + 1e-12f);
  const float* pp = poly + b * 9;
  float s = pp[8];
  #pragma unroll
  for (int k = 7; k >= 0; --k) s = s * rr + pp[k];
  float u = gx * s, v = gy * s;
  float ix = (u + 1.f) * 0.5f * (float)IW - 0.5f;
  float iy = (v + 1.f) * 0.5f * (float)IH - 0.5f;
  float fx = floorf(ix), fy = floorf(iy);
  float wx1 = ix - fx, wy1 = iy - fy;
  int x0i = (int)fx, y0i = (int)fy;
  float w00 = (1.f-wy1)*(1.f-wx1), w01 = (1.f-wy1)*wx1;
  float w10 = wy1*(1.f-wx1),       w11 = wy1*wx1;
  #pragma unroll
  for (int cc = 0; cc < CIMG; ++cc) {
    const float* ip = img + ((size_t)(b*CIMG + cc)) * IH * IW;
    float g00 = ((unsigned)y0i     < IH && (unsigned)x0i     < IW) ? ip[y0i*IW + x0i]         : 0.f;
    float g01 = ((unsigned)y0i     < IH && (unsigned)(x0i+1) < IW) ? ip[y0i*IW + x0i + 1]     : 0.f;
    float g10 = ((unsigned)(y0i+1) < IH && (unsigned)x0i     < IW) ? ip[(y0i+1)*IW + x0i]     : 0.f;
    float g11 = ((unsigned)(y0i+1) < IH && (unsigned)(x0i+1) < IW) ? ip[(y0i+1)*IW + x0i + 1] : 0.f;
    xbev[(((size_t)b*18 + z*3 + cc) * GH + gh) * GW + gw] =
        g00*w00 + g01*w01 + g10*w10 + g11*w11;
  }
}

// ---------------- Stage B: conv0 (18 -> 3, 3x3 SAME) ----------------------
__global__ void k_conv0(const float* __restrict__ xin, const float* __restrict__ w0,
                        const float* __restrict__ b0, float* __restrict__ x0f) {
  int t = blockIdx.x * blockDim.x + threadIdx.x;
  if (t >= NB * 3 * GH * GW) return;
  int x = t % GW; int r = t / GW;
  int y = r % GH; r /= GH;
  int co = r % 3; int b = r / 3;
  float acc = b0[co];
  for (int ci = 0; ci < 18; ++ci)
    #pragma unroll
    for (int ky = 0; ky < 3; ++ky)
      #pragma unroll
      for (int kx = 0; kx < 3; ++kx) {
        int gy = y + ky - 1, gx = x + kx - 1;
        if ((unsigned)gy < GH && (unsigned)gx < GW)
          acc += w0[((co*18 + ci)*3 + ky)*3 + kx] *
                 xin[(((size_t)b*18 + ci)*GH + gy)*GW + gx];
      }
  x0f[t] = acc;
}

// ---------------- Stage C: rb1 conv1 (3->128) + bn1 + relu -> h1 NHWC f16 -
__global__ void k_conv1_rb1(const float* __restrict__ x0f,
                            const float* __restrict__ w1, const float* __restrict__ b1,
                            const float* __restrict__ g,  const float* __restrict__ be,
                            const float* __restrict__ mn, const float* __restrict__ vr,
                            _Float16* __restrict__ h1) {
  int t = blockIdx.x * blockDim.x + threadIdx.x;
  if (t >= NB * GH * GW * 8) return;
  int cg = t & 7; int r = t >> 3;
  int x = r % GW; r /= GW;
  int y = r % GH; int b = r / GH;
  float in[27];
  #pragma unroll
  for (int ky = 0; ky < 3; ++ky)
    #pragma unroll
    for (int kx = 0; kx < 3; ++kx)
      #pragma unroll
      for (int ci = 0; ci < 3; ++ci) {
        int gy = y + ky - 1, gx = x + kx - 1;
        in[(ky*3 + kx)*3 + ci] = ((unsigned)gy < GH && (unsigned)gx < GW)
            ? x0f[(((size_t)b*3 + ci)*GH + gy)*GW + gx] : 0.f;
      }
  for (int j = 0; j < 16; ++j) {
    int co = cg * 16 + j;
    float acc = b1[co];
    #pragma unroll
    for (int tap = 0; tap < 9; ++tap)
      #pragma unroll
      for (int ci = 0; ci < 3; ++ci)
        acc += w1[co*27 + ci*9 + tap] * in[tap*3 + ci];
    float sc = g[co] * rsqrtf(vr[co] + 1e-5f);
    float v = acc * sc + (be[co] - mn[co]*sc);
    v = fmaxf(v, 0.f);
    h1[((size_t)(b*GH + y)*GW + x)*CMID + co] = (_Float16)v;
  }
}

// ---------------- weight transform: rb1.w2 OIHW -> [n][k] f16 -------------
__global__ void k_prep_wt1(const float* __restrict__ w2, _Float16* __restrict__ wt1) {
  int t = blockIdx.x * blockDim.x + threadIdx.x;
  if (t >= CMID * KTOT) return;
  int n = t / KTOT; int k = t - n*KTOT;
  int tap = k >> 7; int ci = k & 127;          // k = tap*128 + cin
  wt1[t] = (_Float16)w2[(n*CMID + ci)*9 + tap];
}

// ---------------- WMMA epilogue helper ------------------------------------
__device__ __forceinline__ void wmma_epilogue(
    v8f acc, int b, int y, int x00, int hi, int nc,
    const float* __restrict__ x0f,
    float sc2, float sh2, float bias,
    float wsa, float wsb, float wsc, float bsv,
    _Float16* __restrict__ r1) {
  #pragma unroll
  for (int j = 0; j < 8; ++j) {
    int x = x00 + j + hi * 8;
    if (x >= GW) continue;
    float s0 = x0f[(((size_t)b*3 + 0)*GH + y)*GW + x];
    float s1 = x0f[(((size_t)b*3 + 1)*GH + y)*GW + x];
    float s2 = x0f[(((size_t)b*3 + 2)*GH + y)*GW + x];
    float res = bsv + wsa*s0 + wsb*s1 + wsc*s2;
    float v = (acc[j] + bias) * sc2 + sh2 + res;
    v = fmaxf(v, 0.f);
    r1[((size_t)(b*GH + y)*GW + x)*CMID + nc] = (_Float16)v;
  }
}

// ---------------- Stage D: rb1 conv2 (128->128 3x3) via WMMA implicit GEMM
// block = 128 thr (4 waves). Wave w computes a 2x2 register tile:
// output rows {y0+2w, y0+2w+1} x out-channel tiles {n0, n0+16}, 16 wide.
// K = 1152 = 36 slices of 32 -> 4 independent WMMAs per slice (144/wave),
// with A fragments shared across the N tiles and B fragments shared across
// the two rows (2 A + 2 B b128-pair loads feed 4 WMMAs).
// Epilogue fuses +b2, bn2, 1x1 shortcut from x0f, relu; writes r1 NHWC f16.
__global__ void __launch_bounds__(128)
k_wmma_conv2_rb1(const _Float16* __restrict__ h1, const _Float16* __restrict__ wt1,
                 const float* __restrict__ x0f,  const float* __restrict__ b2,
                 const float* __restrict__ g,    const float* __restrict__ be,
                 const float* __restrict__ mn,   const float* __restrict__ vr,
                 const float* __restrict__ ws,   const float* __restrict__ bs,
                 _Float16* __restrict__ r1) {
  __shared__ _Float16 sA[10 * 18 * CMID];      // 46080 B activation strip
  const int tid = threadIdx.x;
  const int x00 = blockIdx.x * 16;
  const int y0  = blockIdx.y * 8;
  const int b   = blockIdx.z >> 2;
  const int n0  = (blockIdx.z & 3) * 32;

  // cooperative strip load: rows y0-1..y0+8, cols x00-1..x00+16, 128ch f16
  for (int idx = tid; idx < 2880; idx += 128) {      // 16B chunks
    int c8  = idx & 15;
    int seg = idx >> 4;
    int row = seg / 18;
    int col = seg - row * 18;
    int gy = y0 - 1 + row;
    int gx = x00 - 1 + col;
    f32x4 v = {0.f, 0.f, 0.f, 0.f};
    if ((unsigned)gy < GH && (unsigned)gx < GW)
      v = *(const f32x4*)(h1 + (((size_t)(b*GH + gy)*GW + gx)*CMID + c8*8));
    *(f32x4*)(sA + idx * 8) = v;
  }
  __syncthreads();

  const int lane = tid & 31;
  const int w    = tid >> 5;        // wave id -> rows 2w, 2w+1
  const int mrow = lane & 15;       // A: M row / B,D: N column
  const int hi   = lane >> 4;

  v8f acc00 = {}, acc01 = {}, acc10 = {}, acc11 = {};
  for (int kk = 0; kk < 36; ++kk) {
    int kbase = kk * 32;
    int tap   = kbase >> 7;                 // 32-slice never crosses a tap
    int ky    = tap / 3;
    int kx    = tap - ky * 3;
    int cin0  = (kbase & 127) + hi * 8;
    // A fragments (16x32 f16): elems 0..7 <- K kbase+hi*8.., 8..15 <- +16
    int aoff0 = ((2*w + ky) * 18 + (mrow + kx)) * CMID + cin0;
    int aoff1 = aoff0 + 18 * CMID;          // next output row
    V16U a0, a1;
    a0.f[0] = *(const f32x4*)(sA + aoff0);
    a0.f[1] = *(const f32x4*)(sA + aoff0 + 16);
    a1.f[0] = *(const f32x4*)(sA + aoff1);
    a1.f[1] = *(const f32x4*)(sA + aoff1 + 16);
    // B fragments (32x16 f16): lane&15 = N column, elems 0..7 <- K kbase+hi*8.., 8..15 <- +16
    const _Float16* bp0 = wt1 + (size_t)(n0 + mrow) * KTOT + kbase + hi * 8;
    const _Float16* bp1 = bp0 + (size_t)16 * KTOT;   // next 16 out channels
    V16U b0, b1;
    b0.f[0] = *(const f32x4*)(bp0);
    b0.f[1] = *(const f32x4*)(bp0 + 16);
    b1.f[0] = *(const f32x4*)(bp1);
    b1.f[1] = *(const f32x4*)(bp1 + 16);
    acc00 = __builtin_amdgcn_wmma_f32_16x16x32_f16(false, a0.h, false, b0.h, (short)0, acc00, false, false);
    acc01 = __builtin_amdgcn_wmma_f32_16x16x32_f16(false, a0.h, false, b1.h, (short)0, acc01, false, false);
    acc10 = __builtin_amdgcn_wmma_f32_16x16x32_f16(false, a1.h, false, b0.h, (short)0, acc10, false, false);
    acc11 = __builtin_amdgcn_wmma_f32_16x16x32_f16(false, a1.h, false, b1.h, (short)0, acc11, false, false);
  }

  // per-out-channel fused params for the two N tiles
  float sc2[2], sh2[2], bias[2], wsa[2], wsb[2], wsc[2], bsv[2];
  #pragma unroll
  for (int tle = 0; tle < 2; ++tle) {
    int nc = n0 + 16*tle + mrow;
    sc2[tle]  = g[nc] * rsqrtf(vr[nc] + 1e-5f);
    sh2[tle]  = be[nc] - mn[nc] * sc2[tle];
    bias[tle] = b2[nc];
    wsa[tle]  = ws[nc*3 + 0];
    wsb[tle]  = ws[nc*3 + 1];
    wsc[tle]  = ws[nc*3 + 2];
    bsv[tle]  = bs[nc];
  }
  const int yA = y0 + 2*w;
  const int yB = yA + 1;
  wmma_epilogue(acc00, b, yA, x00, hi, n0 + mrow,      x0f, sc2[0], sh2[0], bias[0], wsa[0], wsb[0], wsc[0], bsv[0], r1);
  wmma_epilogue(acc01, b, yA, x00, hi, n0 + 16 + mrow, x0f, sc2[1], sh2[1], bias[1], wsa[1], wsb[1], wsc[1], bsv[1], r1);
  wmma_epilogue(acc10, b, yB, x00, hi, n0 + mrow,      x0f, sc2[0], sh2[0], bias[0], wsa[0], wsb[0], wsc[0], bsv[0], r1);
  wmma_epilogue(acc11, b, yB, x00, hi, n0 + 16 + mrow, x0f, sc2[1], sh2[1], bias[1], wsa[1], wsb[1], wsc[1], bsv[1], r1);
}

// ---------------- Stage E: rb2 conv1 (128->3) + bn1 + relu -> h2 NHWC f32 -
__global__ void k_conv1_rb2(const _Float16* __restrict__ r1,
                            const float* __restrict__ w,  const float* __restrict__ b1,
                            const float* __restrict__ g,  const float* __restrict__ be,
                            const float* __restrict__ mn, const float* __restrict__ vr,
                            float* __restrict__ h2) {
  __shared__ float sW[9 * CMID * 3];           // [tap][ci][co]
  const int tid = threadIdx.x;
  for (int i = tid; i < 3456; i += 256) {      // from OIHW (3,128,3,3)
    int co  = i / 1152;
    int rem = i - co * 1152;
    int ci  = rem / 9;
    int tap = rem - ci * 9;
    sW[(tap*CMID + ci)*3 + co] = w[i];
  }
  __syncthreads();
  int t = blockIdx.x * 256 + tid;
  if (t >= NB * GH * GW) return;
  int x = t % GW; int r = t / GW;
  int y = r % GH; int b = r / GH;
  float a0 = b1[0], a1 = b1[1], a2 = b1[2];
  for (int ky = 0; ky < 3; ++ky)
    for (int kx = 0; kx < 3; ++kx) {
      int gy = y + ky - 1, gx = x + kx - 1;
      if ((unsigned)gy >= GH || (unsigned)gx >= GW) continue;
      const _Float16* rp = r1 + ((size_t)(b*GH + gy)*GW + gx)*CMID;
      const float* wp = sW + (ky*3 + kx)*CMID*3;
      for (int c8 = 0; c8 < 16; ++c8) {        // 16B vector loads of r1
        H8U u; u.f = *(const f32x4*)(rp + c8*8);
        #pragma unroll
        for (int q = 0; q < 8; ++q) {
          float iv = (float)u.h[q];
          int ci = c8*8 + q;
          a0 += wp[ci*3 + 0] * iv;
          a1 += wp[ci*3 + 1] * iv;
          a2 += wp[ci*3 + 2] * iv;
        }
      }
    }
  float acc[3] = {a0, a1, a2};
  #pragma unroll
  for (int co = 0; co < 3; ++co) {
    float sc = g[co] * rsqrtf(vr[co] + 1e-5f);
    float v = acc[co] * sc + (be[co] - mn[co]*sc);
    h2[(size_t)t*3 + co] = fmaxf(v, 0.f);
  }
}

// ---------------- Stage F: rb2 conv2 + bn2 + 1x1 shortcut (no relu) -------
__global__ void k_final(const float* __restrict__ h2, const _Float16* __restrict__ r1,
                        const float* __restrict__ w2, const float* __restrict__ b2,
                        const float* __restrict__ g,  const float* __restrict__ be,
                        const float* __restrict__ mn, const float* __restrict__ vr,
                        const float* __restrict__ ws, const float* __restrict__ bs,
                        float* __restrict__ out) {
  int t = blockIdx.x * blockDim.x + threadIdx.x;
  if (t >= NB * GH * GW) return;
  int x = t % GW; int r = t / GW;
  int y = r % GH; int b = r / GH;
  float acc[3] = {b2[0], b2[1], b2[2]};
  #pragma unroll
  for (int ky = 0; ky < 3; ++ky)
    #pragma unroll
    for (int kx = 0; kx < 3; ++kx) {
      int gy = y + ky - 1, gx = x + kx - 1;
      if ((unsigned)gy >= GH || (unsigned)gx >= GW) continue;
      const float* hp = h2 + ((size_t)(b*GH + gy)*GW + gx)*3;
      #pragma unroll
      for (int ci = 0; ci < 3; ++ci) {
        float iv = hp[ci];
        #pragma unroll
        for (int co = 0; co < 3; ++co)
          acc[co] += w2[((co*3 + ci)*3 + ky)*3 + kx] * iv;
      }
    }
  float res[3] = {bs[0], bs[1], bs[2]};
  const _Float16* rp = r1 + ((size_t)(b*GH + y)*GW + x)*CMID;
  for (int c8 = 0; c8 < 16; ++c8) {            // 16B vector loads of r1
    H8U u; u.f = *(const f32x4*)(rp + c8*8);
    #pragma unroll
    for (int q = 0; q < 8; ++q) {
      float iv = (float)u.h[q];
      int ci = c8*8 + q;
      #pragma unroll
      for (int co = 0; co < 3; ++co)
        res[co] += ws[co*CMID + ci] * iv;
    }
  }
  #pragma unroll
  for (int co = 0; co < 3; ++co) {
    float sc = g[co] * rsqrtf(vr[co] + 1e-5f);
    float v = acc[co]*sc + (be[co] - mn[co]*sc) + res[co];
    out[(((size_t)b*3 + co)*GH + y)*GW + x] = v;
  }
}

// ---------------- launch ---------------------------------------------------
extern "C" void kernel_launch(void* const* d_in, const int* in_sizes, int n_in,
                              void* d_out, int out_size, void* d_ws, size_t ws_size,
                              hipStream_t stream) {
  // inputs: image_vals, dist_poly, camera_matrix, then params pytree leaves
  // (sorted-dict order): b0, rb1{b1,b2,bn1(g,b,m,v),bn2(g,b,m,v),bs,w1,w2,ws},
  // rb2{...same...}, w0
  const float* img       = (const float*)d_in[0];
  const float* poly      = (const float*)d_in[1];
  const float* cam       = (const float*)d_in[2];
  const float* b0        = (const float*)d_in[3];
  const float* rb1_b1    = (const float*)d_in[4];
  const float* rb1_b2    = (const float*)d_in[5];
  const float* rb1_bn1g  = (const float*)d_in[6];
  const float* rb1_bn1b  = (const float*)d_in[7];
  const float* rb1_bn1m  = (const float*)d_in[8];
  const float* rb1_bn1v  = (const float*)d_in[9];
  const float* rb1_bn2g  = (const float*)d_in[10];
  const float* rb1_bn2b  = (const float*)d_in[11];
  const float* rb1_bn2m  = (const float*)d_in[12];
  const float* rb1_bn2v  = (const float*)d_in[13];
  const float* rb1_bs    = (const float*)d_in[14];
  const float* rb1_w1    = (const float*)d_in[15];
  const float* rb1_w2    = (const float*)d_in[16];
  const float* rb1_ws    = (const float*)d_in[17];
  const float* rb2_b1    = (const float*)d_in[18];
  const float* rb2_b2    = (const float*)d_in[19];
  const float* rb2_bn1g  = (const float*)d_in[20];
  const float* rb2_bn1b  = (const float*)d_in[21];
  const float* rb2_bn1m  = (const float*)d_in[22];
  const float* rb2_bn1v  = (const float*)d_in[23];
  const float* rb2_bn2g  = (const float*)d_in[24];
  const float* rb2_bn2b  = (const float*)d_in[25];
  const float* rb2_bn2m  = (const float*)d_in[26];
  const float* rb2_bn2v  = (const float*)d_in[27];
  const float* rb2_bs    = (const float*)d_in[28];
  const float* rb2_w1    = (const float*)d_in[29];
  const float* rb2_w2    = (const float*)d_in[30];
  const float* rb2_ws    = (const float*)d_in[31];
  const float* w0        = (const float*)d_in[32];

  char* ws = (char*)d_ws;
  float*    xbev = (float*)   (ws + 0);          //  5,760,000 B (2,18,200,200) f32
  float*    x0f  = (float*)   (ws + 5760000);    //    960,000 B (2,3,200,200)  f32
  _Float16* h1   = (_Float16*)(ws + 6720000);    // 20,480,000 B (2,200,200,128) f16
  _Float16* r1   = (_Float16*)(ws + 27200000);   // 20,480,000 B (2,200,200,128) f16
  float*    h2   = (float*)   (ws + 47680000);   //    960,000 B (2,200,200,3)  f32
  _Float16* wt1  = (_Float16*)(ws + 48640000);   //    294,912 B [128][1152]    f16

  k_sample   <<<(NB*NZV*GH*GW + 255)/256, 256, 0, stream>>>(img, poly, cam, xbev);
  k_conv0    <<<(NB*3*GH*GW   + 255)/256, 256, 0, stream>>>(xbev, w0, b0, x0f);
  k_conv1_rb1<<<(NB*GH*GW*8   + 255)/256, 256, 0, stream>>>(
      x0f, rb1_w1, rb1_b1, rb1_bn1g, rb1_bn1b, rb1_bn1m, rb1_bn1v, h1);
  k_prep_wt1 <<<(CMID*KTOT    + 255)/256, 256, 0, stream>>>(rb1_w2, wt1);

  dim3 g4(13, 25, 8);   // xtiles of 16 (padded), ytiles of 8, b*4 + npair
  k_wmma_conv2_rb1<<<g4, 128, 0, stream>>>(
      h1, wt1, x0f, rb1_b2, rb1_bn2g, rb1_bn2b, rb1_bn2m, rb1_bn2v,
      rb1_ws, rb1_bs, r1);

  k_conv1_rb2<<<(NB*GH*GW + 255)/256, 256, 0, stream>>>(
      r1, rb2_w1, rb2_b1, rb2_bn1g, rb2_bn1b, rb2_bn1m, rb2_bn1v, h2);
  k_final    <<<(NB*GH*GW + 255)/256, 256, 0, stream>>>(
      h2, r1, rb2_w2, rb2_b2, rb2_bn2g, rb2_bn2b, rb2_bn2m, rb2_bn2v,
      rb2_ws, rb2_bs, (float*)d_out);

  (void)in_sizes; (void)n_in; (void)out_size; (void)ws_size;
}
